// ir_rgbfuse_67843303408333
// MI455X (gfx1250) — compile-verified, matrix operand order checked by reading, not run on hardware
//
#include <hip/hip_runtime.h>

// ---------------------------------------------------------------------------
// Deformable-conv fusion for MI455X (gfx1250, wave32, WMMA).
// All three conv stages run as f16 WMMA GEMMs with f32 accumulation:
//   offset conv : [16384 x 3456] x [3456 x 32(pad of 18)]
//   mod conv    : [16384 x  192] x [ 192 x 192] (+ sigmoid)
//   deform GEMM : [16384 x 1728] x [1728 x 192]  (fused bilinear gather)
// A-tiles are double-buffered in LDS (one barrier per K-step); B matrices are
// pre-packed into per-lane WMMA fragment order (L2-resident, coalesced b128).
// ---------------------------------------------------------------------------

typedef __attribute__((ext_vector_type(16))) _Float16 v16h;
typedef __attribute__((ext_vector_type(8)))  float    v8f;

#define BATCH  4
#define CH     192
#define HH     64
#define WW     64
#define NPIX   (BATCH * HH * WW)     // 16384

// deform GEMM: K = 9*192 = 1728 -> 54 k-tiles, N = 192 -> 12 n-tiles
#define KT_DC  54
#define NT_DC  12
// mod GEMM: K = 192 -> 6 k-tiles, N = 192 -> 12 n-tiles
#define KT_MOD 6
#define NT_MOD 12
// offset GEMM: K = 9*384 = 3456 -> 108 k-tiles, N = 32 (18 padded) -> 2 tiles
#define KT_OFF 108
#define NT_OFF 2

#define U32_DC  (KT_DC  * NT_DC  * 32 * 8)   // 165888 -> 663552 B
#define U32_MOD (KT_MOD * NT_MOD * 32 * 8)   //  18432 ->  73728 B
#define U32_OFF (KT_OFF * NT_OFF * 32 * 8)   //  55296 -> 221184 B

#define WS_WDC_OFF  0
#define WS_WMOD_OFF 663552
#define WS_WOFF_OFF 737280
#define WS_OFFS_OFF 958464                    // offset map: 4*18*64*64 f32
#define WS_MOD_OFF  2138112                   // mod NHWC: 16384*192 f32
// total ws: 14,721,024 B

// ---------------------------------------------------------------------------
// Shared WMMA fragment helpers.
// A-matrix 16x32 f16 per ISA 7.12.2: lane L holds row M=L%16; lanes 0-15
// carry K {0-7,16-23}, lanes 16-31 carry K {8-15,24-31}; VGPR j = pair (k,k+1).
// B fragments are the same layout applied to B^T (lane holds column N=L%16).
// ---------------------------------------------------------------------------
__device__ __forceinline__ v16h load_a_frag(const _Float16 (*tile)[32], int lane) {
    union { v16h h; unsigned int u[8]; } af;
    const unsigned int* arow = (const unsigned int*)&tile[lane & 15][0];
    const int bk2 = (lane >> 4) * 4;
    #pragma unroll
    for (int j = 0; j < 4; ++j) {
        af.u[j]     = arow[bk2 + j];
        af.u[j + 4] = arow[8 + bk2 + j];
    }
    return af.h;
}

__device__ __forceinline__ v16h load_b_frag(const unsigned int* __restrict__ wfrag,
                                            int kt, int NT, int nt, int lane) {
    union { v16h h; unsigned int u[8]; } bf;
    const unsigned int* bp = &wfrag[(((size_t)kt * NT + nt) * 32 + lane) * 8];
    #pragma unroll
    for (int j = 0; j < 8; ++j) bf.u[j] = bp[j];
    return bf.h;
}

__device__ __forceinline__ v8f wmma_f16(v16h a, v16h b, v8f c) {
    return __builtin_amdgcn_wmma_f32_16x16x32_f16(
        false, a, false, b, (short)0, c, false, false);
}

__device__ __forceinline__ void frag_decode(int t, int NT,
                                            int& kt, int& nt, int& lane, int& j) {
    j = t & 7; int r = t >> 3;
    lane = r & 31; r >>= 5;
    nt = r % NT; kt = r / NT;
}

__device__ __forceinline__ unsigned int pack2h(float v0, float v1) {
    union { _Float16 h[2]; unsigned int u; } pk;
    pk.h[0] = (_Float16)v0; pk.h[1] = (_Float16)v1;
    return pk.u;
}

__device__ __forceinline__ int frag_kwin(int lane, int j) {
    int bk = (lane >> 4) * 8;
    return (j < 4) ? (bk + 2 * j) : (16 + bk + 2 * (j - 4));
}

// ---------------------------------------------------------------------------
// Pack kernels: B^T[n][k] -> per-lane fragment order.
// ---------------------------------------------------------------------------
__global__ void pack_wdc(const float* __restrict__ w_dc,
                         unsigned int* __restrict__ wfrag) {
    int t = blockIdx.x * blockDim.x + threadIdx.x;
    if (t >= U32_DC) return;
    int kt, nt, lane, j;  frag_decode(t, NT_DC, kt, nt, lane, j);
    int kk = kt / 6, cg = kt % 6;              // K = kk*192 + c (tap-major)
    int ky = kk / 3, kx = kk % 3;
    int n = nt * 16 + (lane & 15);
    int c = cg * 32 + frag_kwin(lane, j);
    wfrag[t] = pack2h(w_dc[((n * CH + c    ) * 3 + ky) * 3 + kx],
                      w_dc[((n * CH + c + 1) * 3 + ky) * 3 + kx]);
}

__global__ void pack_wmod(const float* __restrict__ w_mod,
                          unsigned int* __restrict__ wfrag) {
    int t = blockIdx.x * blockDim.x + threadIdx.x;
    if (t >= U32_MOD) return;
    int kt, nt, lane, j;  frag_decode(t, NT_MOD, kt, nt, lane, j);
    int n = nt * 16 + (lane & 15);
    int c = kt * 32 + frag_kwin(lane, j);      // K = c (1x1 conv)
    wfrag[t] = pack2h(w_mod[n * CH + c], w_mod[n * CH + c + 1]);
}

__global__ void pack_woff(const float* __restrict__ w_off,
                          unsigned int* __restrict__ wfrag) {
    int t = blockIdx.x * blockDim.x + threadIdx.x;
    if (t >= U32_OFF) return;
    int kt, nt, lane, j;  frag_decode(t, NT_OFF, kt, nt, lane, j);
    int n = nt * 16 + (lane & 15);             // N padded: 18 real, 32 total
    int tap = kt / 12;                         // K = tap*384 + c
    int c = (kt % 12) * 32 + frag_kwin(lane, j);
    float v0 = 0.f, v1 = 0.f;
    if (n < 18) {
        v0 = w_off[(n * 384 + c    ) * 9 + tap];
        v1 = w_off[(n * 384 + c + 1) * 9 + tap];
    }
    wfrag[t] = pack2h(v0, v1);
}

// ---------------------------------------------------------------------------
// Kernel: offset conv as WMMA GEMM.  Block = 16 pixels x 64 threads (2 waves,
// one n-tile each).  K-loop: 108 tiles (tap fixed per tile, 32 channels).
// The x-vs-irf source split is uniform per k-tile (192 % 32 == 0), and the
// halo handling is branch-free: clamped-address load * {0,1} mask.
// ---------------------------------------------------------------------------
__global__ void __launch_bounds__(64) off_wmma(
        const float* __restrict__ x, const float* __restrict__ irf,
        const unsigned int* __restrict__ wfrag, const float* __restrict__ b_off,
        float* __restrict__ offs) {
    __shared__ _Float16 Atile[2][16][32];
    const int tid = threadIdx.x, lane = tid & 31, wave = tid >> 5;
    const int p0 = blockIdx.x * 16;
    const int b = p0 >> 12, rem = p0 & 4095, y = rem >> 6, x0 = rem & 63;

    v8f acc;
    #pragma unroll
    for (int r = 0; r < 8; ++r) acc[r] = 0.f;

    for (int kt = 0; kt < KT_OFF; ++kt) {
        const int buf = kt & 1;
        const int tap = kt / 12, cg = (kt % 12) * 32;
        const int tx = tap % 3;
        const int yy = y - 1 + tap / 3;
        const int yi = min(max(yy, 0), HH - 1);
        const float ymask = (yy >= 0 && yy < HH) ? 1.f : 0.f;
        const bool from_x = (cg < CH);          // uniform scalar branch
        #pragma unroll
        for (int i = 0; i < 8; ++i) {
            int e = tid + 64 * i;
            int p = e >> 5, cl = e & 31;
            int xx = x0 + p - 1 + tx;
            int xi = min(max(xx, 0), WW - 1);
            float mask = (xx >= 0 && xx < WW) ? ymask : 0.f;
            float v;
            if (from_x) {
                v = x[((b * HH + yi) * WW + xi) * CH + cg + cl];
            } else {
                v = irf[((b * CH + (cg - CH + cl)) * HH + yi) * WW + xi];
            }
            Atile[buf][p][cl] = (_Float16)(v * mask);
        }
        __syncthreads();
        v16h af = load_a_frag(Atile[buf], lane);
        v16h bf = load_b_frag(wfrag, kt, NT_OFF, wave, lane);
        acc = wmma_f16(af, bf, acc);
    }

    const int n = wave * 16 + (lane & 15);
    if (n < 18) {
        const float bias = b_off[n];
        #pragma unroll
        for (int r = 0; r < 8; ++r) {
            int m = (lane >> 4) * 8 + r;
            offs[((b * 18 + n) * HH + y) * WW + (x0 + m)] = acc[r] + bias;
        }
    }
}

// ---------------------------------------------------------------------------
// Kernel: 1x1 mod conv as WMMA GEMM + sigmoid.  Block = 16 pixels x 128
// threads (4 waves x 3 n-tiles).  K-loop: 6 tiles of 32 channels.
// Output NHWC so the deform epilogue reads coalesced.
// ---------------------------------------------------------------------------
__global__ void __launch_bounds__(128) mod_wmma(
        const float* __restrict__ irf, const unsigned int* __restrict__ wfrag,
        const float* __restrict__ b_mod, float* __restrict__ modw) {
    __shared__ _Float16 Atile[2][16][32];
    const int tid = threadIdx.x, lane = tid & 31, wave = tid >> 5;
    const int p0 = blockIdx.x * 16;
    const int b = p0 >> 12, rem = p0 & 4095, y = rem >> 6, x0 = rem & 63;

    v8f acc[3];
    #pragma unroll
    for (int t = 0; t < 3; ++t)
        #pragma unroll
        for (int r = 0; r < 8; ++r) acc[t][r] = 0.f;

    for (int kt = 0; kt < KT_MOD; ++kt) {
        const int buf = kt & 1;
        // lanes vary pixel fastest: irf row [c][x0..x0+15] is contiguous
        #pragma unroll
        for (int i = 0; i < 4; ++i) {
            int e = tid + 128 * i;
            int p = e & 15, cl = e >> 4;
            int c = kt * 32 + cl;
            Atile[buf][p][cl] =
                (_Float16)irf[((b * CH + c) * HH + y) * WW + x0 + p];
        }
        __syncthreads();
        v16h af = load_a_frag(Atile[buf], lane);
        #pragma unroll
        for (int t = 0; t < 3; ++t) {
            v16h bf = load_b_frag(wfrag, kt, NT_MOD, wave * 3 + t, lane);
            acc[t] = wmma_f16(af, bf, acc[t]);
        }
    }

    #pragma unroll
    for (int t = 0; t < 3; ++t) {
        const int n = (wave * 3 + t) * 16 + (lane & 15);
        const float bias = b_mod[n];
        #pragma unroll
        for (int r = 0; r < 8; ++r) {
            int m = (lane >> 4) * 8 + r;
            int pg = p0 + m;
            modw[(size_t)pg * CH + n] = 1.f / (1.f + expf(-(acc[t][r] + bias)));
        }
    }
}

// ---------------------------------------------------------------------------
// Kernel: fused bilinear-gather + WMMA GEMM + bias + mod-multiply.
// Block = 16 pixels x 128 threads (4 waves x 3 n-tiles).  54 K-tiles,
// double-buffered A-tile -> one barrier per K-step.
// ---------------------------------------------------------------------------
__global__ void __launch_bounds__(128) deform_wmma(
        const float* __restrict__ x, const float* __restrict__ offs,
        const unsigned int* __restrict__ wfrag, const float* __restrict__ b_dc,
        const float* __restrict__ modw, float* __restrict__ out) {
    __shared__ _Float16 Atile[2][16][32];
    __shared__ int   nidx[16][9][4];
    __shared__ float nwt [16][9][4];

    const int tid = threadIdx.x, lane = tid & 31, wave = tid >> 5;
    const int p0 = blockIdx.x * 16;
    const int b = p0 >> 12, rem = p0 & 4095, y = rem >> 6, x0 = rem & 63;

    // ---- precompute bilinear corner indices/weights for 16 px x 9 taps ----
    for (int item = tid; item < 144; item += 128) {
        int p = item / 9, kk = item % 9;
        int xw = x0 + p;
        float dy = offs[((b * 18 + 2 * kk    ) * HH + y) * WW + xw];
        float dx = offs[((b * 18 + 2 * kk + 1) * HH + y) * WW + xw];
        float py = (float)(y  - 1 + kk / 3) + dy;
        float px = (float)(xw - 1 + kk % 3) + dx;
        float y0f = floorf(py), x0f = floorf(px);
        float wy1 = py - y0f, wx1 = px - x0f;
        float wy0 = 1.f - wy1, wx0 = 1.f - wx1;
        #pragma unroll
        for (int jy = 0; jy < 2; ++jy) {
            #pragma unroll
            for (int jx = 0; jx < 2; ++jx) {
                float yf = y0f + (float)jy, xf = x0f + (float)jx;
                bool valid = (yf >= 0.f) && (yf <= (float)(HH - 1)) &&
                             (xf >= 0.f) && (xf <= (float)(WW - 1));
                int yi = min(max((int)yf, 0), HH - 1);
                int xi = min(max((int)xf, 0), WW - 1);
                float wgt = (jy ? wy1 : wy0) * (jx ? wx1 : wx0);
                int j = jy * 2 + jx;
                nidx[p][kk][j] = ((b * HH + yi) * WW + xi) * CH;  // NHWC base
                nwt [p][kk][j] = valid ? wgt : 0.f;
            }
        }
    }
    __syncthreads();

    v8f acc[3];
    #pragma unroll
    for (int t = 0; t < 3; ++t)
        #pragma unroll
        for (int r = 0; r < 8; ++r) acc[t][r] = 0.f;

    for (int kt = 0; kt < KT_DC; ++kt) {
        const int buf = kt & 1;
        const int kk = kt / 6, cg = (kt % 6) * 32;

        // build 16x32 f16 A-tile: coalesced 32-channel gathers (NHWC x)
        #pragma unroll
        for (int i = 0; i < 4; ++i) {
            int e = tid + 128 * i;
            int p = e >> 5, cl = e & 31;
            int cglob = cg + cl;
            const int*   id = nidx[p][kk];
            const float* wt = nwt [p][kk];
            float v = wt[0] * x[id[0] + cglob] + wt[1] * x[id[1] + cglob]
                    + wt[2] * x[id[2] + cglob] + wt[3] * x[id[3] + cglob];
            Atile[buf][p][cl] = (_Float16)v;
        }
        __syncthreads();

        v16h af = load_a_frag(Atile[buf], lane);
        if (kt + 1 < KT_DC)
            __builtin_prefetch(
                &wfrag[(((size_t)(kt + 1) * NT_DC + wave * 3) * 32 + lane) * 8],
                0, 0);
        #pragma unroll
        for (int t = 0; t < 3; ++t) {
            v16h bf = load_b_frag(wfrag, kt, NT_DC, wave * 3 + t, lane);
            acc[t] = wmma_f16(af, bf, acc[t]);
        }
    }

    // ---- epilogue: + bias, * mod, store NHWC f32 ----
    #pragma unroll
    for (int t = 0; t < 3; ++t) {
        const int n = (wave * 3 + t) * 16 + (lane & 15);
        const float bias = b_dc[n];
        #pragma unroll
        for (int r = 0; r < 8; ++r) {
            int m  = (lane >> 4) * 8 + r;
            int pg = p0 + m;
            out[(size_t)pg * CH + n] =
                (acc[t][r] + bias) * modw[(size_t)pg * CH + n];
        }
    }
}

// ---------------------------------------------------------------------------
extern "C" void kernel_launch(void* const* d_in, const int* in_sizes, int n_in,
                              void* d_out, int out_size, void* d_ws, size_t ws_size,
                              hipStream_t stream) {
    const float* x     = (const float*)d_in[0];   // [4,64,64,192] NHWC
    const float* irf   = (const float*)d_in[1];   // [4,192,64,64] NCHW
    const float* w_off = (const float*)d_in[2];   // [18,384,3,3]
    const float* b_off = (const float*)d_in[3];   // [18]
    const float* w_mod = (const float*)d_in[4];   // [192,192,1,1]
    const float* b_mod = (const float*)d_in[5];   // [192]
    const float* w_dc  = (const float*)d_in[6];   // [192,192,3,3]
    const float* b_dc  = (const float*)d_in[7];   // [192]
    float* out = (float*)d_out;                   // [4,64,64,192] NHWC f32

    char* ws = (char*)d_ws;                       // needs ~14.8 MB
    unsigned int* wfdc  = (unsigned int*)(ws + WS_WDC_OFF);
    unsigned int* wfmod = (unsigned int*)(ws + WS_WMOD_OFF);
    unsigned int* wfoff = (unsigned int*)(ws + WS_WOFF_OFF);
    float*        offsb = (float*)(ws + WS_OFFS_OFF);
    float*        modw  = (float*)(ws + WS_MOD_OFF);

    pack_wdc  <<<(U32_DC  + 255) / 256, 256, 0, stream>>>(w_dc,  wfdc);
    pack_wmod <<<(U32_MOD + 255) / 256, 256, 0, stream>>>(w_mod, wfmod);
    pack_woff <<<(U32_OFF + 255) / 256, 256, 0, stream>>>(w_off, wfoff);

    off_wmma  <<<NPIX / 16,  64, 0, stream>>>(x, irf, wfoff, b_off, offsb);
    mod_wmma  <<<NPIX / 16, 128, 0, stream>>>(irf, wfmod, b_mod, modw);
    deform_wmma<<<NPIX / 16, 128, 0, stream>>>(x, offsb, wfdc, b_dc, modw, out);
}
